// DeformConv2d_3539053052120
// MI455X (gfx1250) — matrix operand order repaired and checked by reading, not compiled, not run on hardware
//
#include <hip/hip_runtime.h>
#include <hip/hip_bf16.h>

// Problem constants (match reference)
#define NB   8
#define HH   64
#define WW   64
#define CC   256
#define HWSZ 4096      // H*W
#define GG   4
#define GD_  64        // C/G
#define KK   9         // KS*KS
#define OMD  112       // OM_DIM

typedef __attribute__((ext_vector_type(16))) __bf16 v16bf;
typedef __attribute__((ext_vector_type(8)))  float  v8f;

union Frag {
    v16bf v;
    unsigned int u[8];
    uint4 q[2];
};

__device__ __forceinline__ unsigned int pack2_bf16(float lo, float hi) {
    // round-to-nearest-even f32 -> bf16, packed into one dword
    unsigned int a = __float_as_uint(lo);
    unsigned int b = __float_as_uint(hi);
    a = (a + 0x7FFFu + ((a >> 16) & 1u)) >> 16;
    b = (b + 0x7FFFu + ((b >> 16) & 1u)) >> 16;
    return a | (b << 16);
}

// A-fragment layout (16-bit A 16x32, ISA 7.12.2):
//   Apack[mt][ks][lane][v] (8 dwords per lane)
//   lane&15 -> M row in tile; lane>>4 -> K-offset half (0 or +8)
//   v 0..3 -> K = kb+2v,2v+1 ; v 4..7 -> K = kb+16+2(v-4),...  (kb = ks*32 + half*8)
__device__ __forceinline__ size_t frag_idx(int m, int k0, int ksteps) {
    int mt = m >> 4, r = m & 15;
    int ks = k0 >> 5, kk = k0 & 31;
    int q  = kk >> 3;                         // which 8-wide K block
    int h  = q & 1;                           // lane half
    int v  = ((q >> 1) << 2) + ((kk & 7) >> 1);
    int lane = (h << 4) | r;
    return ((size_t)(mt * ksteps + ks) * 32 + lane) * 8 + v;
}

// ---------------------------------------------------------------------------
// Pack B = W^T (W is [Ncols x Kc] row-major f32) into bf16 B-fragments.
//   lane&15 -> N column; lane>>4 -> K-offset half (0 or +8); v 0..3 -> K=kb+2v,2v+1 ; v 4..7 -> K=kb+16+2(v-4),...
// ---------------------------------------------------------------------------
__global__ __launch_bounds__(256) void pack_b_kernel(
    const float* __restrict__ W, unsigned int* __restrict__ Bpack,
    int Ncols, int Kc)
{
    int ksteps = Kc >> 5;
    int total  = (Ncols >> 4) * ksteps * 32 * 8;
    int idx = blockIdx.x * blockDim.x + threadIdx.x;
    if (idx >= total) return;
    int v    = idx & 7;
    int lane = (idx >> 3) & 31;
    int t    = idx >> 8;              // nt*ksteps + ks
    int nt   = t / ksteps;
    int ks   = t - nt * ksteps;
    int n    = nt * 16 + (lane & 15);
    int k0   = ks * 32 + (lane >> 4) * 8 + ((v >> 2) << 4) + 2 * (v & 3);
    size_t base = (size_t)n * Kc + k0;
    Bpack[idx] = pack2_bf16(W[base], W[base + 1]);
}

// ---------------------------------------------------------------------------
// Bulk pack of an f32 activation matrix [M x Kc] into bf16 A-fragments.
// ---------------------------------------------------------------------------
__global__ __launch_bounds__(256) void pack_a_kernel(
    const float* __restrict__ A, unsigned int* __restrict__ Apack,
    int M, int Kc)
{
    int ksteps = Kc >> 5;
    int total  = (M >> 4) * ksteps * 32 * 8;
    int idx = blockIdx.x * blockDim.x + threadIdx.x;
    if (idx >= total) return;
    int v    = idx & 7;
    int lane = (idx >> 3) & 31;
    int t    = idx >> 8;              // mt*ksteps + ks
    int mt   = t / ksteps;
    int ks   = t - mt * ksteps;
    int kb   = ks * 32 + (lane >> 4) * 8;
    int k0   = (v < 4) ? (kb + 2 * v) : (kb + 16 + 2 * (v - 4));
    int m    = mt * 16 + (lane & 15);
    size_t base = (size_t)m * Kc + k0;
    Apack[idx] = pack2_bf16(A[base], A[base + 1]);
}

// ---------------------------------------------------------------------------
// GEMM: C[M x Ncols] = A*B + bias. A, B pre-packed bf16 fragments.
// One wave computes a 16 x (16*NT) strip. K fully unrolled; fragments are
// double-buffered so step s+1's loads issue before step s's WMMAs.
// ---------------------------------------------------------------------------
template<int NT, int KSTEPS>
__global__ __launch_bounds__(256) void wmma_gemm_kernel(
    const unsigned int* __restrict__ Apack,
    const unsigned int* __restrict__ Bpack,
    const float* __restrict__ bias, float* __restrict__ C,
    int M, int Ncols)
{
    const int ntiles  = Ncols >> 4;
    const int ngroups = ntiles / NT;
    const int mtiles  = M >> 4;
    int wave = (blockIdx.x * blockDim.x + threadIdx.x) >> 5;
    int lane = threadIdx.x & 31;
    if (wave >= mtiles * ngroups) return;
    int mt = wave / ngroups;
    int ng = wave - mt * ngroups;
    int row  = lane & 15;
    int half = lane >> 4;

    // Two base pointers; every load below is a constant offset from these.
    const uint4* ab = (const uint4*)(Apack + ((size_t)(mt * KSTEPS) * 32 + lane) * 8);
    const uint4* bb = (const uint4*)(Bpack + ((size_t)(ng * NT * KSTEPS) * 32 + lane) * 8);

    v8f acc[NT] = {};
    Frag fa_cur, fb_cur[NT];

    // prologue: load k-step 0
    fa_cur.q[0] = ab[0];
    fa_cur.q[1] = ab[1];
    #pragma unroll
    for (int nt = 0; nt < NT; ++nt) {
        fb_cur[nt].q[0] = bb[nt * (KSTEPS * 64) + 0];
        fb_cur[nt].q[1] = bb[nt * (KSTEPS * 64) + 1];
    }

    #pragma unroll
    for (int ks = 0; ks < KSTEPS; ++ks) {
        Frag fa_nxt, fb_nxt[NT];
        if (ks + 1 < KSTEPS) {
            fa_nxt.q[0] = ab[(ks + 1) * 64 + 0];            // +1024B per step
            fa_nxt.q[1] = ab[(ks + 1) * 64 + 1];
            #pragma unroll
            for (int nt = 0; nt < NT; ++nt) {
                fb_nxt[nt].q[0] = bb[nt * (KSTEPS * 64) + (ks + 1) * 64 + 0];
                fb_nxt[nt].q[1] = bb[nt * (KSTEPS * 64) + (ks + 1) * 64 + 1];
            }
        }
        #pragma unroll
        for (int nt = 0; nt < NT; ++nt)
            acc[nt] = __builtin_amdgcn_wmma_f32_16x16x32_bf16(
                false, fa_cur.v, false, fb_cur[nt].v, (short)0, acc[nt],
                false, false);
        if (ks + 1 < KSTEPS) {
            fa_cur = fa_nxt;
            #pragma unroll
            for (int nt = 0; nt < NT; ++nt)
                fb_cur[nt] = fb_nxt[nt];
        }
    }

    int mbase = mt * 16 + half * 8;     // C layout: VGPR i -> M = mbase + i
    #pragma unroll
    for (int nt = 0; nt < NT; ++nt) {
        int n = (ng * NT + nt) * 16 + row;
        float bv = bias[n];
        #pragma unroll
        for (int i = 0; i < 8; ++i)
            C[(size_t)(mbase + i) * Ncols + n] = acc[nt][i] + bv;
    }
}

// ---------------------------------------------------------------------------
// Depthwise 3x3 conv (pad=1) -> writes bf16 A-fragments directly.
// Each thread computes channels (c0, c0+1) of one pixel = one packed dword.
// ---------------------------------------------------------------------------
__global__ __launch_bounds__(256) void dwconv_pack_kernel(
    const float* __restrict__ x, const float* __restrict__ wdw,
    const float* __restrict__ bdw, unsigned int* __restrict__ Apack)
{
    int idx = blockIdx.x * blockDim.x + threadIdx.x;
    if (idx >= NB * HWSZ * (CC / 2)) return;
    int c0 = (idx & (CC / 2 - 1)) * 2;
    int p  = (idx >> 7) & (HWSZ - 1);
    int n  = idx >> 19;
    int y  = p >> 6, xw = p & 63;
    float a0 = bdw[c0], a1 = bdw[c0 + 1];
    #pragma unroll
    for (int ky = 0; ky < 3; ++ky) {
        int yy = y + ky - 1;
        if (yy < 0 || yy >= HH) continue;
        #pragma unroll
        for (int kx = 0; kx < 3; ++kx) {
            int xx = xw + kx - 1;
            if (xx < 0 || xx >= WW) continue;
            const float2 xv = *(const float2*)&x[((size_t)n * HWSZ + yy * WW + xx) * CC + c0];
            const float2 wv = *(const float2*)&wdw[(ky * 3 + kx) * CC + c0];
            a0 += xv.x * wv.x;
            a1 += xv.y * wv.y;
        }
    }
    int m = n * HWSZ + p;
    Apack[frag_idx(m, c0, CC >> 5)] = pack2_bf16(a0, a1);
}

// ---------------------------------------------------------------------------
// Deformable bilinear sampling -> writes bf16 A-fragments directly.
// Each thread computes channels (c0, c0+1) of one (n,p,g) = one packed dword.
// xg = projected input [N, H*W, C] viewed [N, H*W, G, GD]; om = [N, H*W, OMD]
// ---------------------------------------------------------------------------
__global__ __launch_bounds__(256) void deform_sample_pack_kernel(
    const float* __restrict__ xg, const float* __restrict__ om,
    unsigned int* __restrict__ Apack)
{
    int idx = blockIdx.x * blockDim.x + threadIdx.x;
    if (idx >= NB * HWSZ * (CC / 2)) return;
    int c0 = (idx & (GD_ / 2 - 1)) * 2;
    int g  = (idx >> 5) & (GG - 1);
    int p  = (idx >> 7) & (HWSZ - 1);
    int n  = idx >> 19;
    int y  = p >> 6, xw = p & 63;

    const float* omp = om + ((size_t)n * HWSZ + p) * OMD;
    float a0 = 0.0f, a1 = 0.0f;
    #pragma unroll
    for (int k = 0; k < KK; ++k) {
        float off0 = omp[(g * KK + k) * 2 + 0];
        float off1 = omp[(g * KK + k) * 2 + 1];
        float msk  = omp[2 * GG * KK + g * KK + k];
        float px = (float)xw + (float)(k % 3 - 1) + off0;
        float py = (float)y  + (float)(k / 3 - 1) + off1;
        float x0 = floorf(px), y0 = floorf(py);
        #pragma unroll
        for (int dy = 0; dy < 2; ++dy) {
            #pragma unroll
            for (int dx = 0; dx < 2; ++dx) {
                float xi = x0 + (float)dx;
                float yi = y0 + (float)dy;
                float wgt = (1.0f - fabsf(px - xi)) * (1.0f - fabsf(py - yi));
                bool valid = (xi >= 0.0f) & (xi <= (float)(WW - 1)) &
                             (yi >= 0.0f) & (yi <= (float)(HH - 1));
                wgt = valid ? wgt * msk : 0.0f;
                int xc = min(max((int)xi, 0), WW - 1);
                int yc = min(max((int)yi, 0), HH - 1);
                const float2 v = *(const float2*)&xg[
                    (((size_t)n * HWSZ + yc * WW + xc) * GG + g) * GD_ + c0];
                a0 += wgt * v.x;
                a1 += wgt * v.y;
            }
        }
    }
    int m = n * HWSZ + p;
    Apack[frag_idx(m, g * GD_ + c0, CC >> 5)] = pack2_bf16(a0, a1);
}

// ---------------------------------------------------------------------------
extern "C" void kernel_launch(void* const* d_in, const int* in_sizes, int n_in,
                              void* d_out, int out_size, void* d_ws, size_t ws_size,
                              hipStream_t stream)
{
    (void)in_sizes; (void)n_in; (void)out_size; (void)ws_size;
    const float* x    = (const float*)d_in[0];
    const float* w_in = (const float*)d_in[3];
    const float* b_in = (const float*)d_in[4];
    const float* w_out= (const float*)d_in[5];
    const float* b_out= (const float*)d_in[6];
    const float* w_dw = (const float*)d_in[7];
    const float* b_dw = (const float*)d_in[8];
    const float* w_pw = (const float*)d_in[9];
    const float* b_pw = (const float*)d_in[10];
    float* out = (float*)d_out;

    const int M = NB * HWSZ;                                 // 32768
    // Workspace carve-up
    char* ws = (char*)d_ws;
    float* buf_out = (float*)ws;                             // M*CC  f32
    float* buf_om  = buf_out + (size_t)M * CC;               // M*OMD f32
    unsigned int* slot0 = (unsigned int*)(buf_om + (size_t)M * OMD); // M*CC/2 dwords
    unsigned int* slot1 = slot0 + (size_t)M * CC / 2;        // M*CC/2 dwords
    unsigned int* pin   = slot1 + (size_t)M * CC / 2;        // 32768 dwords
    unsigned int* ppw   = pin + 16 * 8 * 256;                // 14336 dwords
    unsigned int* pout  = ppw + 7 * 8 * 256;                 // 32768 dwords

    // 1) weight fragments
    pack_b_kernel<<<(16 * 8 * 256) / 256, 256, 0, stream>>>(w_in,  pin,  CC,  CC);
    pack_b_kernel<<<( 7 * 8 * 256) / 256, 256, 0, stream>>>(w_pw,  ppw,  OMD, CC);
    pack_b_kernel<<<(16 * 8 * 256) / 256, 256, 0, stream>>>(w_out, pout, CC,  CC);

    // 2) xA fragments (slot0)
    pack_a_kernel<<<((M >> 4) * 8 * 256) / 256, 256, 0, stream>>>(x, slot0, M, CC);

    // 3) out = x @ w_in^T + b_in           (f32 result needed by sampler)
    wmma_gemm_kernel<4, 8><<<((M >> 4) * 4 * 32) / 256, 256, 0, stream>>>(
        slot0, pin, b_in, buf_out, M, CC);

    // 4) dw fragments = depthwise3x3(x) + b_dw   (slot1)
    dwconv_pack_kernel<<<(NB * HWSZ * (CC / 2)) / 256, 256, 0, stream>>>(
        x, w_dw, b_dw, slot1);

    // 5) om = dw @ w_pw^T + b_pw
    wmma_gemm_kernel<7, 8><<<((M >> 4) * 1 * 32) / 256, 256, 0, stream>>>(
        slot1, ppw, b_pw, buf_om, M, OMD);

    // 6) samp fragments (slot0 reused; x fragments dead after step 3)
    deform_sample_pack_kernel<<<(NB * HWSZ * (CC / 2)) / 256, 256, 0, stream>>>(
        buf_out, buf_om, slot0);

    // 7) result = samp @ w_out^T + b_out
    wmma_gemm_kernel<4, 8><<<((M >> 4) * 4 * 32) / 256, 256, 0, stream>>>(
        slot0, pout, b_out, out, M, CC);
}
